// Int8Conv_20641612824710
// MI455X (gfx1250) — matrix operand order repaired and checked by reading, not compiled, not run on hardware
//
#include <hip/hip_runtime.h>

typedef __attribute__((ext_vector_type(16))) _Float16 v16h;
typedef __attribute__((ext_vector_type(8)))  _Float16 v8h;
typedef __attribute__((ext_vector_type(8)))  float    v8f;
typedef __attribute__((ext_vector_type(4)))  float    v4f;

// Problem dims (fixed by the reference)
constexpr int Bn  = 64;
constexpr int H   = 112;
constexpr int W   = 112;
constexpr int CIN = 32;
constexpr int F   = 32;

// Tiling
constexpr int SEG        = 64;          // output pixels per workgroup (4 M-tiles of 16)
constexpr int HALO_W     = SEG + 2;     // 66 staged input pixels (left/right halo)
constexpr int XPITCH     = 40;          // halves per staged pixel (32 + 8 pad -> bank-conflict free)
constexpr int WFRAG_H    = 24;          // halves per lane per B-fragment (16 + 8 pad)
constexpr int WFRAG_STR  = 32 * WFRAG_H;  // 768 halves per fragment
constexpr int NFRAG      = 9 * 2;       // 9 taps x 2 N-tiles
constexpr int WS_HALVES  = NFRAG * WFRAG_STR; // 13824 halves = 27648 B in d_ws

// ---------------------------------------------------------------------------
// Prep: swizzle int8-valued weights [3,3,CIN,F] (int32) into per-lane f16
// WMMA B-fragments. B-matrix (32x16 f16) per-lane layout (ISA 7.12.2):
//   lane L: n = L&15 ; element e (VGPR e/2, half e%2): k = e<8 ? (L>>4)*8 + e : 16 + (L>>4)*8 + e-8
// int8 values are exactly representable in f16; `scale` is applied at epilogue.
// ---------------------------------------------------------------------------
__global__ void Int8Conv_prep_weights(const int* __restrict__ w,
                                      _Float16* __restrict__ wsW) {
    int tid = blockIdx.x * blockDim.x + threadIdx.x;
    if (tid >= NFRAG * 32 * 16) return;
    int e  = tid & 15;
    int l  = (tid >> 4) & 31;
    int fr = tid >> 9;            // fragment: t*2 + ntile
    int t  = fr >> 1;             // tap 0..8
    int nt = fr & 1;              // N-tile
    int k  = (e < 8) ? ((l >> 4) << 3) + e : 16 + ((l >> 4) << 3) + (e - 8); // input channel
    int n  = nt * 16 + (l & 15);  // output channel
    int wv = w[(t * CIN + k) * F + n];
    wsW[fr * WFRAG_STR + l * WFRAG_H + e] = (_Float16)wv;
}

// ---------------------------------------------------------------------------
// Main conv: one workgroup = one (b,h) row x 64-pixel segment x 32 channels.
// 8 waves: wave = (mtile 0..3, ntile 0..1), each does 9 WMMAs (one per tap).
// ---------------------------------------------------------------------------
__global__ __launch_bounds__(256)
void Int8Conv_20641612824710_kernel(const float* __restrict__ x,
                                    const _Float16* __restrict__ wsW,
                                    const float* __restrict__ scale,
                                    float* __restrict__ out) {
    __shared__ _Float16 sX[3 * HALO_W * XPITCH]; // 15840 B staged input (f16)
    __shared__ _Float16 sW[WS_HALVES];           // 27648 B B-fragments

    const int tid = threadIdx.x;
    const int bh  = blockIdx.y;
    const int b   = bh / H;
    const int h   = bh % H;
    const int w0  = blockIdx.x * SEG;

    // ---- copy pre-swizzled weight fragments into LDS (dword copies) ----
    {
        const uint32_t* src = (const uint32_t*)wsW;
        uint32_t*       dst = (uint32_t*)sW;
        for (int i = tid; i < WS_HALVES / 2; i += 256) dst[i] = src[i];
    }

    // ---- stage x rows h-1..h+1, pixels w0-1..w0+64, fp32 -> f16 ----
    const float* xrow = x + (size_t)(b * H) * W * CIN;
    // prefetch next row of this segment (CDNA5 global_prefetch_b8)
    {
        int hp = (h + 2 < H) ? h + 2 : h;
        __builtin_prefetch(&xrow[((size_t)hp * W + w0) * CIN], 0, 3);
    }
    for (int i = tid; i < 3 * HALO_W * (CIN / 4); i += 256) {
        int c4  = (i & 7) << 2;          // channel group (CIN/4 == 8)
        int p   = (i >> 3) % HALO_W;     // staged pixel 0..65
        int kh  = (i >> 3) / HALO_W;     // 0..2
        int hin = h + kh - 1;
        int win = w0 + p - 1;
        v4f v = {0.0f, 0.0f, 0.0f, 0.0f};
        if ((unsigned)hin < (unsigned)H && (unsigned)win < (unsigned)W)
            v = *(const v4f*)&xrow[((size_t)hin * W + win) * CIN + c4];
        _Float16* d = &sX[(kh * HALO_W + p) * XPITCH + c4];
        d[0] = (_Float16)v.x; d[1] = (_Float16)v.y;
        d[2] = (_Float16)v.z; d[3] = (_Float16)v.w;
    }
    __syncthreads();

    // ---- per-wave implicit GEMM ----
    const int wv   = tid >> 5;       // wave 0..7
    const int lane = tid & 31;
    const int mt   = wv >> 1;        // M-tile 0..3
    const int nt   = wv & 1;         // N-tile 0..1
    const int m0   = mt * 16;
    if (w0 + m0 >= W) return;        // partial segment (w0=64 has 3 M-tiles)

    const int lm = lane & 15;        // A: row m ; C/D: column n
    const int lh = lane >> 4;        // lane-half selects K-group / M-half

    v8f acc = {0.f, 0.f, 0.f, 0.f, 0.f, 0.f, 0.f, 0.f};

#pragma unroll
    for (int t = 0; t < 9; ++t) {
        const int kh = t / 3, kw = t % 3;
        // A fragment: 16 px x 32 cin. lane: m = lm, halves:
        //   e<8  -> K = lh*8 + e        (chunk at +lh*8)
        //   e>=8 -> K = 16 + lh*8 + e-8 (chunk at +16+lh*8)
        const int lp    = m0 + lm + kw;                       // staged pixel (+kw-1+1)
        const int abase = (kh * HALO_W + lp) * XPITCH + lh * 8;
        v8h a0 = *(const v8h*)&sX[abase];
        v8h a1 = *(const v8h*)&sX[abase + 16];
        v16h a = __builtin_shufflevector(a0, a1, 0, 1, 2, 3, 4, 5, 6, 7,
                                                 8, 9, 10, 11, 12, 13, 14, 15);
        // B fragment: pre-swizzled, contiguous 16 halves per lane
        const int wbase = (t * 2 + nt) * WFRAG_STR + lane * WFRAG_H;
        v8h b0 = *(const v8h*)&sW[wbase];
        v8h b1 = *(const v8h*)&sW[wbase + 8];
        v16h bm = __builtin_shufflevector(b0, b1, 0, 1, 2, 3, 4, 5, 6, 7,
                                                  8, 9, 10, 11, 12, 13, 14, 15);
        acc = __builtin_amdgcn_wmma_f32_16x16x32_f16(
            /*neg_a=*/false, a, /*neg_b=*/false, bm,
            /*c_mod=*/(short)0, acc, /*reuse_a=*/false, /*reuse_b=*/false);
    }

    // ---- epilogue: per-channel dequant scale, f32 store ----
    // C/D layout: VGPR r -> m = lh*8 + r ; n = lm
    const int   ch    = nt * 16 + lm;
    const float sc    = scale[ch];
    float* obase = out + (((size_t)b * H + h) * W + (w0 + m0)) * F + ch;
#pragma unroll
    for (int r = 0; r < 8; ++r) {
        int m = lh * 8 + r;
        obase[(size_t)m * F] = acc[r] * sc;
    }
}

extern "C" void kernel_launch(void* const* d_in, const int* in_sizes, int n_in,
                              void* d_out, int out_size, void* d_ws, size_t ws_size,
                              hipStream_t stream) {
    const float* x     = (const float*)d_in[0];
    const int*   w     = (const int*)d_in[1];
    const float* scale = (const float*)d_in[2];
    float*       out   = (float*)d_out;
    _Float16*    wsW   = (_Float16*)d_ws;   // 27648 B of scratch used

    // 1) swizzle weights into WMMA B-fragment layout (f16, exact for int8)
    Int8Conv_prep_weights<<<(NFRAG * 32 * 16 + 255) / 256, 256, 0, stream>>>(w, wsW);

    // 2) conv: grid = (W segments, B*H rows)
    dim3 grid((W + SEG - 1) / SEG, Bn * H);
    Int8Conv_20641612824710_kernel<<<grid, 256, 0, stream>>>(x, wsW, scale, out);
}
